// Conv2d_Xnor_ReAct_87419764342822
// MI455X (gfx1250) — 8-bit matrix operand order repaired and checked by reading, compiled, not run on hardware
//
#include <hip/hip_runtime.h>
#include <hip/hip_bf16.h>

typedef int v4i __attribute__((ext_vector_type(4)));
typedef int v8i __attribute__((ext_vector_type(8)));

#define C_IN   256
#define C_OUT  256
#define HH     56
#define WW     56
#define BB     32
#define HP     58
#define WP     58
#define NTOT   (BB * HH * WW)
#define KBLK   36
#define ACT_BYTES ((size_t)BB * HP * WP * C_IN)
#define PW_BYTES  ((size_t)16 * KBLK * 32 * 32 * 1)

__global__ void pack_weights(const float* __restrict__ w, signed char* __restrict__ pw) {
    int idx = blockIdx.x * 256 + threadIdx.x;
    int byte = idx & 31;
    int lane = (idx >> 5) & 31;
    int blk  = idx >> 10;
    int kb   = blk % KBLK;
    int mblk = blk / KBLK;
    int o = mblk * 16 + (lane & 15);
    int klocal = ((byte >> 4) << 5) + (byte & 15) + ((lane >> 4) << 4);
    int k = kb * 64 + klocal;
    int tap = k >> 8;
    int c   = k & 255;
    int kh = tap / 3, kw = tap % 3;
    float wv = w[(((size_t)o * C_IN + c) * 3 + kh) * 3 + kw];
    pw[idx] = (wv > 0.0f) ? (signed char)1 : (signed char)-1;
}

__global__ void binarize_act(const float* __restrict__ x, const float* __restrict__ alpha,
                             signed char* __restrict__ act) {
    int pid = blockIdx.x * blockDim.x + threadIdx.x;
    if (pid >= BB * HP * WP) return;
    int xx = pid % WP;
    int t  = pid / WP;
    int yy = t % HP;
    int b  = t / HP;
    signed char* dst = act + (size_t)pid * C_IN;
    bool inside = (yy >= 1) && (yy <= HH) && (xx >= 1) && (xx <= WW);
    int y = yy - 1, xc = xx - 1;
    const float* src = x + (((size_t)b * C_IN) * HH + y) * WW + xc;

    for (int cb = 0; cb < 4; ++cb) {
        int buf[16];
        signed char* bb = (signed char*)buf;
        if (inside) {
            #pragma unroll
            for (int j = 0; j < 64; ++j) {
                int c = cb * 64 + j;
                float xv = src[(size_t)c * (HH * WW)];
                bb[j] = ((xv - alpha[c]) > 0.0f) ? (signed char)1 : (signed char)-1;
            }
        } else {
            #pragma unroll
            for (int j = 0; j < 16; ++j) buf[j] = 0;
        }
        v4i* d4 = (v4i*)(dst + cb * 64);
        const v4i* s4 = (const v4i*)buf;
        d4[0] = s4[0]; d4[1] = s4[1]; d4[2] = s4[2]; d4[3] = s4[3];
    }
}

__global__ __launch_bounds__(256) void xnor_gemm(const signed char* __restrict__ act,
                                                 const signed char* __restrict__ pw,
                                                 float* __restrict__ out) {
    int lane = threadIdx.x & 31;
    int wave = threadIdx.x >> 5;
    int m0 = (wave & 3) * 64;
    int n0 = blockIdx.x * 64 + (wave >> 2) * 32;
    int half = lane >> 4;
    int l15  = lane & 15;

    const signed char* bbase[2];
    int bN[2], bP[2];
    #pragma unroll
    for (int t = 0; t < 2; ++t) {
        int col = n0 + t * 16 + l15;
        int b = col / (HH * WW);
        int p = col % (HH * WW);
        int y = p / WW;
        int xc = p % WW;
        bbase[t] = act + (((size_t)(b * HP + y) * WP + xc) * C_IN) + half * 16;
        bN[t] = b; bP[t] = p;
    }

    const signed char* abase = pw + (size_t)(m0 >> 4) * KBLK * 1024 + (size_t)lane * 32;

    v8i acc[4][2] = {};

    #pragma unroll
    for (int kb = 0; kb < KBLK; ++kb) {
        const int tap = kb >> 2;
        const int kh = tap / 3, kw = tap % 3;
        const int boff = ((kh * WP + kw) << 8) + ((kb & 3) << 6);

        v8i bfrag[2];
        #pragma unroll
        for (int t = 0; t < 2; ++t) {
            v4i lo = *(const v4i*)(bbase[t] + boff);
            v4i hi = *(const v4i*)(bbase[t] + boff + 32);
            bfrag[t] = __builtin_shufflevector(lo, hi, 0, 1, 2, 3, 4, 5, 6, 7);
        }
        v8i afrag[4];
        #pragma unroll
        for (int i = 0; i < 4; ++i) {
            const signed char* ap = abase + (size_t)(i * KBLK + kb) * 1024;
            v4i alo = *(const v4i*)(ap);
            v4i ahi = *(const v4i*)(ap + 16);
            afrag[i] = __builtin_shufflevector(alo, ahi, 0, 1, 2, 3, 4, 5, 6, 7);
        }
        #pragma unroll
        for (int i = 0; i < 4; ++i) {
            #pragma unroll
            for (int t = 0; t < 2; ++t) {
                acc[i][t] = __builtin_amdgcn_wmma_i32_16x16x64_iu8(
                    true, afrag[i], true, bfrag[t], acc[i][t], false, false);
            }
        }
    }

    #pragma unroll
    for (int i = 0; i < 4; ++i) {
        int rbase = m0 + i * 16 + half * 8;
        #pragma unroll
        for (int t = 0; t < 2; ++t) {
            float* op = out + (size_t)bN[t] * (C_OUT * HH * WW)
                            + (size_t)rbase * (HH * WW) + bP[t];
            #pragma unroll
            for (int j = 0; j < 8; ++j)
                op[(size_t)j * (HH * WW)] = (float)acc[i][t][j];
        }
    }
}

extern "C" void kernel_launch(void* const* d_in, const int* in_sizes, int n_in,
                              void* d_out, int out_size, void* d_ws, size_t ws_size,
                              hipStream_t stream) {
    const float* x     = (const float*)d_in[0];
    const float* alpha = (const float*)d_in[1];
    const float* w     = (const float*)d_in[2];
    float* out = (float*)d_out;

    signed char* act = (signed char*)d_ws;
    signed char* pw  = (signed char*)d_ws + ACT_BYTES;

    binarize_act<<<(BB * HP * WP + 255) / 256, 256, 0, stream>>>(x, alpha, act);
    pack_weights<<<(int)(PW_BYTES / 256), 256, 0, stream>>>(w, pw);
    xnor_gemm<<<NTOT / 64, 256, 0, stream>>>(act, pw, out);
}
